// GPTLanguageModel_37134287241768
// MI455X (gfx1250) — matrix operand order repaired and checked by reading, not compiled, not run on hardware
//
#include <hip/hip_runtime.h>
#include <hip/hip_bf16.h>
#include <cstdint>

// ---------------- model constants ----------------
enum : int { V = 32000, E = 1024, H = 16, L = 6, F = 4096, S = 1024, Bb = 4, T = 1024, D = 64 };
enum : int { NR = Bb * T };           // 4096 token rows
static constexpr float EPS = 1e-5f;

// ---------------- WMMA types ----------------
typedef __attribute__((ext_vector_type(16))) __bf16 v16bf;
typedef __attribute__((ext_vector_type(8)))  __bf16 v8bf;
typedef __attribute__((ext_vector_type(8)))  float  v8f;

__device__ __forceinline__ v8f wmma_bf16(v16bf a, v16bf b, v8f c) {
  return __builtin_amdgcn_wmma_f32_16x16x32_bf16(false, a, false, b, (short)0, c, false, false);
}

// A-fragment, 16x32 bf16 (ISA 7.12.2): lane m=l&15; runs K=(l<16?0:8)+0..7 and 16+(l<16?0:8)+0..7
__device__ __forceinline__ v16bf load_afrag(const __bf16* p, int ld) {
  int lane = threadIdx.x & 31;
  int m    = lane & 15;
  int koff = (lane < 16) ? 0 : 8;
  const __bf16* r = p + m * ld;
  v16bf a;
  *(v8bf*)&a       = *(const v8bf*)(r + koff);
  *((v8bf*)&a + 1) = *(const v8bf*)(r + 16 + koff);
  return a;
}

// B-fragment, 32x16 bf16 from N-major storage BT[n][k]: lane n=l&15; runs K=(l<16?0:8)+0..7 and 16+(l<16?0:8)+0..7
__device__ __forceinline__ v16bf load_bfrag(const __bf16* p, int ld) {
  int lane = threadIdx.x & 31;
  int n    = lane & 15;
  int koff = (lane < 16) ? 0 : 8;
  const __bf16* r = p + n * ld + koff;
  v16bf b;
  *(v8bf*)&b       = *(const v8bf*)(r);
  *((v8bf*)&b + 1) = *(const v8bf*)(r + 16);
  return b;
}

// CDNA5 async global->LDS copy (ASYNCcnt-tracked; INST_OFFSET applies to both addresses)
__device__ __forceinline__ void async_copy16(const __bf16* lds, const __bf16* g) {
  uint32_t l = (uint32_t)(uintptr_t)lds;           // flat aperture low 32 bits = LDS byte address
  uint64_t a = (uint64_t)(uintptr_t)g;
  asm volatile("global_load_async_to_lds_b128 %0, %1, off"
               :: "v"(l), "v"(a) : "memory");
}
__device__ __forceinline__ void async_copy32(const __bf16* lds, const __bf16* g) {
  uint32_t l = (uint32_t)(uintptr_t)lds;
  uint64_t a = (uint64_t)(uintptr_t)g;
  asm volatile("global_load_async_to_lds_b128 %0, %1, off\n\t"
               "global_load_async_to_lds_b128 %0, %1, off offset:16"
               :: "v"(l), "v"(a) : "memory");
}
__device__ __forceinline__ void wait_async0() {
  asm volatile("s_wait_asynccnt 0x0" ::: "memory");
}

// DPP ROW_XMASK (0x160|mask) reductions within each 16-lane half: pure VALU, no LDS.
__device__ __forceinline__ float dpp_xor1(float x) {
  return __int_as_float(__builtin_amdgcn_update_dpp(0, __float_as_int(x), 0x161, 0xf, 0xf, true));
}
__device__ __forceinline__ float dpp_xor2(float x) {
  return __int_as_float(__builtin_amdgcn_update_dpp(0, __float_as_int(x), 0x162, 0xf, 0xf, true));
}
__device__ __forceinline__ float dpp_xor4(float x) {
  return __int_as_float(__builtin_amdgcn_update_dpp(0, __float_as_int(x), 0x164, 0xf, 0xf, true));
}
__device__ __forceinline__ float dpp_xor8(float x) {
  return __int_as_float(__builtin_amdgcn_update_dpp(0, __float_as_int(x), 0x168, 0xf, 0xf, true));
}
__device__ __forceinline__ float half_reduce_max(float x) {
  x = fmaxf(x, dpp_xor1(x)); x = fmaxf(x, dpp_xor2(x));
  x = fmaxf(x, dpp_xor4(x)); x = fmaxf(x, dpp_xor8(x));
  return x;
}
__device__ __forceinline__ float half_reduce_sum(float x) {
  x += dpp_xor1(x); x += dpp_xor2(x); x += dpp_xor4(x); x += dpp_xor8(x);
  return x;
}

// ---------------- conversion kernels ----------------
__global__ void conv_bf16(const float* __restrict__ in, __bf16* __restrict__ out, long long n) {
  long long i = (long long)blockIdx.x * blockDim.x + threadIdx.x;
  if (i < n) out[i] = (__bf16)in[i];
}

// in: K x N row-major fp32  ->  out: N x K row-major bf16 (i.e. transposed)
__global__ void convT_bf16(const float* __restrict__ in, __bf16* __restrict__ out, int K, int N) {
  long long i = (long long)blockIdx.x * blockDim.x + threadIdx.x;
  long long tot = (long long)K * N;
  if (i < tot) {
    int k = (int)(i / N), n = (int)(i % N);
    out[(long long)n * K + k] = (__bf16)in[i];
  }
}

// ---------------- embedding ----------------
__global__ void embed_kernel(const int* __restrict__ idx, const float* __restrict__ wte,
                             const float* __restrict__ wpe, float* __restrict__ x) {
  int row = blockIdx.x;                 // 0..NR-1  (= b*T + t)
  int t   = row & (T - 1);
  long long v = idx[row];
  for (int i = threadIdx.x; i < E; i += blockDim.x)
    x[(long long)row * E + i] = wte[v * E + i] + wpe[(long long)t * E + i];
}

// ---------------- layernorm (fp32 in -> bf16 out) ----------------
__launch_bounds__(256)
__global__ void layernorm_bf16(const float* __restrict__ x, const float* __restrict__ g,
                               const float* __restrict__ b, __bf16* __restrict__ out) {
  __shared__ float red[256];
  int row = blockIdx.x, tid = threadIdx.x;
  const float* xr = x + (long long)row * E;
  float s = 0.f;
  for (int i = tid; i < E; i += 256) s += xr[i];
  red[tid] = s; __syncthreads();
  for (int o = 128; o > 0; o >>= 1) { if (tid < o) red[tid] += red[tid + o]; __syncthreads(); }
  float mean = red[0] * (1.0f / E);
  __syncthreads();
  float vv = 0.f;
  for (int i = tid; i < E; i += 256) { float d = xr[i] - mean; vv += d * d; }
  red[tid] = vv; __syncthreads();
  for (int o = 128; o > 0; o >>= 1) { if (tid < o) red[tid] += red[tid + o]; __syncthreads(); }
  float rstd = rsqrtf(red[0] * (1.0f / E) + EPS);
  for (int i = tid; i < E; i += 256)
    out[(long long)row * E + i] = (__bf16)((xr[i] - mean) * rstd * g[i] + b[i]);
}

// ---------------- bf16 WMMA GEMM ----------------
// C(MxN,f32 acc) = A(MxK bf16, row-major) * BT(NxK bf16).
// Double-buffered LDS (padded stride 40 halfs to spread banks), async global->LDS staging,
// one barrier per K step. Epilogue behavior is compile-time (no runtime branches).
// Requires M%128==0, N%128==0, K%32==0.
template <bool BIAS, bool ACT, bool RES, bool OUTF, bool OUTB>
__launch_bounds__(256)
__global__ void gemm_bf16(const __bf16* __restrict__ A, const __bf16* __restrict__ BT,
                          const float* __restrict__ bias, const float* __restrict__ res,
                          float* __restrict__ outF, __bf16* __restrict__ outB,
                          int M, int N, int K) {
  constexpr int LDP = 40;                               // padded LDS row stride (halfs)
  __shared__ __align__(16) __bf16 As[2][128 * LDP];
  __shared__ __align__(16) __bf16 Bs[2][128 * LDP];
  int tid = threadIdx.x, wid = tid >> 5, lane = tid & 31;
  int wr = wid >> 1, wc = wid & 1;                      // 4x2 wave grid
  int bm = blockIdx.y * 128, bn = blockIdx.x * 128;

  v8f acc[2][4];
#pragma unroll
  for (int mi = 0; mi < 2; ++mi)
#pragma unroll
    for (int ni = 0; ni < 4; ++ni) acc[mi][ni] = (v8f)0.f;

  int lrow = tid >> 1;
  int lkh  = (tid & 1) * 16;
  const __bf16* agBase = A  + (long long)(bm + lrow) * K + lkh;
  const __bf16* bgBase = BT + (long long)(bn + lrow) * K + lkh;

  auto stage = [&](int k0, int buf) {
    async_copy32(&As[buf][lrow * LDP + lkh], agBase + k0);
    async_copy32(&Bs[buf][lrow * LDP + lkh], bgBase + k0);
  };

  const int nsteps = K >> 5;
  stage(0, 0);
  for (int s = 0; s < nsteps; ++s) {
    int buf = s & 1;
    wait_async0();            // data for buf (issued last step) has landed
    __syncthreads();          // all waves waited; buf^1 readers (step s-1) are done
    if (s + 1 < nsteps) stage((s + 1) << 5, buf ^ 1);

    v16bf af[2], bfr[4];
#pragma unroll
    for (int mi = 0; mi < 2; ++mi) af[mi]  = load_afrag(&As[buf][(wr * 32 + mi * 16) * LDP], LDP);
#pragma unroll
    for (int ni = 0; ni < 4; ++ni) bfr[ni] = load_bfrag(&Bs[buf][(wc * 64 + ni * 16) * LDP], LDP);
#pragma unroll
    for (int mi = 0; mi < 2; ++mi)
#pragma unroll
      for (int ni = 0; ni < 4; ++ni)
        acc[mi][ni] = wmma_bf16(af[mi], bfr[ni], acc[mi][ni]);
  }

  // epilogue: C layout row = v + 8*(lane>=16), col = lane&15
  int cn = lane & 15, rb = (lane >> 4) * 8;
#pragma unroll
  for (int mi = 0; mi < 2; ++mi)
#pragma unroll
    for (int ni = 0; ni < 4; ++ni) {
      int col = bn + wc * 64 + ni * 16 + cn;
      float bv = BIAS ? bias[col] : 0.f;
#pragma unroll
      for (int v = 0; v < 8; ++v) {
        int row = bm + wr * 32 + mi * 16 + rb + v;
        float xv = acc[mi][ni][v] + bv;
        if (ACT) xv = 0.5f * xv * (1.0f + erff(xv * 0.70710678118654752f));  // exact GELU
        long long off = (long long)row * N + col;
        if (RES)  xv += res[off];
        if (OUTF) outF[off] = xv;
        if (OUTB) outB[off] = (__bf16)xv;
      }
    }
}

// ---------------- flash attention (causal, bf16 in/out, fp32 online softmax) ----------------
// grid: (T/128, H, B); block 256 = 8 waves, each wave owns 16 query rows.
// Row max reduced per chunk via DPP ROW_XMASK; softmax denominator kept lane-partial
// (rescale factor is half-uniform) and reduced once at the end.
__launch_bounds__(256)
__global__ void attn_kernel(const __bf16* __restrict__ Q, const __bf16* __restrict__ Kb,
                            const __bf16* __restrict__ Vb, __bf16* __restrict__ O) {
  constexpr int KLD = 72;   // Ks row stride (halfs), padded
  constexpr int VLD = 40;   // Vs row stride
  constexpr int PLD = 40;   // Ps row stride
  __shared__ __align__(16) __bf16 Ks[32 * KLD];       // [key][d]
  __shared__ __align__(16) __bf16 Vs[64 * VLD];       // [d][key]  (transposed)
  __shared__ __align__(16) __bf16 Ps[8][16 * PLD];    // per-wave P tile [m][key]

  int tid = threadIdx.x, wid = tid >> 5, lane = tid & 31;
  int qt = blockIdx.x, h = blockIdx.y, b = blockIdx.z;
  int qrow0 = qt * 128 + wid * 16;
  long long rowbase = (long long)b * T;
  int hcol = h * D;

  // Q fragments (two K=32 steps covering D=64)
  v16bf qf[2];
  {
    int m = lane & 15, koff = (lane < 16) ? 0 : 8;
#pragma unroll
    for (int ds = 0; ds < 2; ++ds) {
      const __bf16* r = Q + (rowbase + qrow0 + m) * E + hcol + ds * 32;
      *(v8bf*)&qf[ds]       = *(const v8bf*)(r + koff);
      *((v8bf*)&qf[ds] + 1) = *(const v8bf*)(r + 16 + koff);
    }
  }

  float mrow[8], lrow[8];                 // lrow is lane-partial
  v8f oacc[4];
#pragma unroll
  for (int v = 0; v < 8; ++v) { mrow[v] = -1e30f; lrow[v] = 0.f; }
#pragma unroll
  for (int nb = 0; nb < 4; ++nb) oacc[nb] = (v8f)0.f;

  const float scale = 0.125f;                 // 1/sqrt(64)
  int cn = lane & 15, rb = (lane >> 4) * 8;
  int kend = qt * 128 + 128;
  int skey = tid >> 3;                        // staging: key row 0..31
  int sd8  = (tid & 7) * 8;                   // staging: d offset 0,8,..,56

  for (int kc = 0; kc < kend; kc += 32) {
    __syncthreads();                          // previous chunk's frag reads complete
    // stage K asynchronously (one b128 per thread), V transposed via VGPRs
    async_copy16(&Ks[skey * KLD + sd8], Kb + (rowbase + kc + skey) * E + hcol + sd8);
    {
      const __bf16* vg = Vb + (rowbase + kc + skey) * E + hcol + sd8;
      v8bf vv = *(const v8bf*)vg;
#pragma unroll
      for (int j = 0; j < 8; ++j) Vs[(sd8 + j) * VLD + skey] = vv[j];
    }
    wait_async0();
    __syncthreads();

    // S = Q * K^T : two 16x16 key subtiles
    v8f s0 = (v8f)0.f, s1 = (v8f)0.f;
#pragma unroll
    for (int ds = 0; ds < 2; ++ds) {
      s0 = wmma_bf16(qf[ds], load_bfrag(&Ks[0 * 16 * KLD + ds * 32], KLD), s0);
      s1 = wmma_bf16(qf[ds], load_bfrag(&Ks[1 * 16 * KLD + ds * 32], KLD), s1);
    }

    // online softmax, per row (v, lane-half); max via DPP, denominator lane-partial
#pragma unroll
    for (int v = 0; v < 8; ++v) {
      int qabs = qrow0 + rb + v;
      float a0 = s0[v] * scale, a1 = s1[v] * scale;
      if (kc + cn      > qabs) a0 = -1e30f;
      if (kc + 16 + cn > qabs) a1 = -1e30f;
      float cm = half_reduce_max(fmaxf(a0, a1));
      float nm = fmaxf(mrow[v], cm);
      float sc = __expf(mrow[v] - nm);          // uniform across the 16-lane half
      float p0 = __expf(a0 - nm), p1 = __expf(a1 - nm);
      lrow[v] = lrow[v] * sc + (p0 + p1);       // lane-partial denominator
      mrow[v] = nm;
#pragma unroll
      for (int nb = 0; nb < 4; ++nb) oacc[nb][v] *= sc;
      Ps[wid][(rb + v) * PLD + cn]      = (__bf16)p0;
      Ps[wid][(rb + v) * PLD + 16 + cn] = (__bf16)p1;
    }
    // wave-private LDS RAW: wait on DS counter before reading P back
    asm volatile("s_wait_dscnt 0x0" ::: "memory");

    // O += P * V (K=32 over keys, 4 d-subtiles)
    v16bf pf = load_afrag(&Ps[wid][0], PLD);
#pragma unroll
    for (int nb = 0; nb < 4; ++nb)
      oacc[nb] = wmma_bf16(pf, load_bfrag(&Vs[nb * 16 * VLD], VLD), oacc[nb]);
  }

  // normalize (reduce lane-partial denominator once) and store
#pragma unroll
  for (int v = 0; v < 8; ++v) {
    float inv = 1.0f / half_reduce_sum(lrow[v]);
    long long row = rowbase + qrow0 + rb + v;
#pragma unroll
    for (int nb = 0; nb < 4; ++nb)
      O[row * E + hcol + nb * 16 + cn] = (__bf16)(oacc[nb][v] * inv);
  }
}

// ---------------- launcher ----------------
extern "C" void kernel_launch(void* const* d_in, const int* in_sizes, int n_in,
                              void* d_out, int out_size, void* d_ws, size_t ws_size,
                              hipStream_t stream) {
  const int*   idx   = (const int*)  d_in[0];
  const float* wte   = (const float*)d_in[1];
  const float* wpe   = (const float*)d_in[2];
  const float* Wq    = (const float*)d_in[3];
  const float* Wk    = (const float*)d_in[4];
  const float* Wv    = (const float*)d_in[5];
  const float* Wo    = (const float*)d_in[6];
  const float* ln1g  = (const float*)d_in[7];
  const float* ln1b  = (const float*)d_in[8];
  const float* ln2g  = (const float*)d_in[9];
  const float* ln2b  = (const float*)d_in[10];
  const float* W1    = (const float*)d_in[11];
  const float* b1    = (const float*)d_in[12];
  const float* W2    = (const float*)d_in[13];
  const float* b2    = (const float*)d_in[14];
  const float* lnfg  = (const float*)d_in[15];
  const float* lnfb  = (const float*)d_in[16];
  float* logits = (float*)d_out;

  // carve workspace (256B aligned)
  char* w = (char*)d_ws;
  auto carve = [&](size_t bytes) -> void* {
    void* p = (void*)w;
    w += (bytes + 255) & ~(size_t)255;
    return p;
  };
  __bf16* wteB = (__bf16*)carve((size_t)V * E * 2);
  __bf16* WqT  = (__bf16*)carve((size_t)E * E * 2);
  __bf16* WkT  = (__bf16*)carve((size_t)E * E * 2);
  __bf16* WvT  = (__bf16*)carve((size_t)E * E * 2);
  __bf16* WoT  = (__bf16*)carve((size_t)E * E * 2);
  __bf16* W1T  = (__bf16*)carve((size_t)E * F * 2);
  __bf16* W2T  = (__bf16*)carve((size_t)F * E * 2);
  float*  x    = (float*) carve((size_t)NR * E * 4);
  __bf16* hB   = (__bf16*)carve((size_t)NR * E * 2);
  __bf16* qB   = (__bf16*)carve((size_t)NR * E * 2);
  __bf16* kB   = (__bf16*)carve((size_t)NR * E * 2);
  __bf16* vB   = (__bf16*)carve((size_t)NR * E * 2);
  __bf16* oB   = (__bf16*)carve((size_t)NR * E * 2);
  __bf16* m1   = (__bf16*)carve((size_t)NR * F * 2);

  auto blocks1d = [](long long n) { return (unsigned)((n + 255) / 256); };

  // tied embedding table to bf16 (used for logits)
  conv_bf16<<<blocks1d((long long)V * E), 256, 0, stream>>>(wte, wteB, (long long)V * E);
  // token + position embedding
  embed_kernel<<<NR, 256, 0, stream>>>(idx, wte, wpe, x);

  dim3 gE (E / 128, NR / 128);
  dim3 gF (F / 128, NR / 128);
  dim3 gV (V / 128, NR / 128);
  dim3 gAt(T / 128, H, Bb);

  for (int l = 0; l < L; ++l) {
    long long oEE = (long long)l * E * E, oEF = (long long)l * E * F;
    convT_bf16<<<blocks1d((long long)E * E), 256, 0, stream>>>(Wq + oEE, WqT, E, E);
    convT_bf16<<<blocks1d((long long)E * E), 256, 0, stream>>>(Wk + oEE, WkT, E, E);
    convT_bf16<<<blocks1d((long long)E * E), 256, 0, stream>>>(Wv + oEE, WvT, E, E);
    convT_bf16<<<blocks1d((long long)E * E), 256, 0, stream>>>(Wo + oEE, WoT, E, E);
    convT_bf16<<<blocks1d((long long)E * F), 256, 0, stream>>>(W1 + oEF, W1T, E, F);
    convT_bf16<<<blocks1d((long long)E * F), 256, 0, stream>>>(W2 + oEF, W2T, F, E);

    layernorm_bf16<<<NR, 256, 0, stream>>>(x, ln1g + l * E, ln1b + l * E, hB);
    // q/k/v: bf16 out only
    gemm_bf16<false,false,false,false,true><<<gE, 256, 0, stream>>>(hB, WqT, nullptr, nullptr, nullptr, qB, NR, E, E);
    gemm_bf16<false,false,false,false,true><<<gE, 256, 0, stream>>>(hB, WkT, nullptr, nullptr, nullptr, kB, NR, E, E);
    gemm_bf16<false,false,false,false,true><<<gE, 256, 0, stream>>>(hB, WvT, nullptr, nullptr, nullptr, vB, NR, E, E);

    attn_kernel<<<gAt, 256, 0, stream>>>(qB, kB, vB, oB);

    // x += o @ Wo  (residual, fp32 out)
    gemm_bf16<false,false,true,true,false><<<gE, 256, 0, stream>>>(oB, WoT, nullptr, x, x, nullptr, NR, E, E);

    layernorm_bf16<<<NR, 256, 0, stream>>>(x, ln2g + l * E, ln2b + l * E, hB);
    // m1 = gelu(h @ W1 + b1): bias + act + bf16 out
    gemm_bf16<true,true,false,false,true><<<gF, 256, 0, stream>>>(hB, W1T, b1 + (long long)l * F, nullptr, nullptr, m1, NR, F, E);
    // x += m1 @ W2 + b2: bias + residual + fp32 out
    gemm_bf16<true,false,true,true,false><<<gE, 256, 0, stream>>>(m1, W2T, b2 + (long long)l * E, x, x, nullptr, NR, E, F);
  }

  layernorm_bf16<<<NR, 256, 0, stream>>>(x, lnfg, lnfb, hB);
  // logits = h @ wte^T   (wte is V x E == already N-major over K=E): fp32 out only
  gemm_bf16<false,false,false,true,false><<<gV, 256, 0, stream>>>(hB, wteB, nullptr, nullptr, logits, nullptr, NR, V, E);
}
